// MoE_87479893885667
// MI455X (gfx1250) — matrix operand order repaired and checked by reading, not compiled, not run on hardware
//
#include <hip/hip_runtime.h>
#include <hip/hip_bf16.h>
#include <math.h>

#define DIMD   2048
#define INTERD 1408
#define NE     16
#define NPSE   18          // 16 routed + 2 shared pseudo-experts
#define TOPKD  4
#define TTOK   2048
#define ROWS_ROUTED (TTOK * TOPKD)          // 8192
#define ROWS_TOTAL  (ROWS_ROUTED + 2*TTOK)  // 12288

typedef __bf16 bfvec16 __attribute__((ext_vector_type(16)));
typedef __bf16 bfvec8  __attribute__((ext_vector_type(8)));
typedef float  v8f     __attribute__((ext_vector_type(8)));
typedef int    v4i     __attribute__((vector_size(16)));   // matches builtin proto

#if defined(__has_builtin)
#if __has_builtin(__builtin_amdgcn_global_load_async_to_lds_b128) && \
    __has_builtin(__builtin_amdgcn_s_wait_asynccnt)
#define USE_ASYNC_LDS 1
#endif
#endif

// ---------- helpers ----------
__device__ __forceinline__ unsigned short f2bf(float f) {
    union { __bf16 b; unsigned short u; } v;
    v.b = (__bf16)f;                          // hardware cvt, RNE
    return v.u;
}

__device__ __forceinline__ float fast_sigmoid(float x) {
    return __builtin_amdgcn_rcpf(1.0f + __expf(-x));
}

// A fragment: 16x32 bf16, LDS row-major [rows][32].
// lane 0-15: M=lane, K in {kb..kb+7, kb+16..kb+23}, kb=0 ; lanes 16-31: kb=8.
__device__ __forceinline__ bfvec16 load_a_frag(const unsigned short* buf, int m_off, int lane) {
    int m  = m_off + (lane & 15);
    int kb = (lane >> 4) * 8;
    bfvec8 lo = *(const bfvec8*)(buf + m * 32 + kb);
    bfvec8 hi = *(const bfvec8*)(buf + m * 32 + kb + 16);
    return __builtin_shufflevector(lo, hi, 0,1,2,3,4,5,6,7,8,9,10,11,12,13,14,15);
}

// B fragment: 32x16 bf16 (KxN), LDS stored transposed [n][32k].
// lane 0-15: N=lane, K in {kb..kb+7, kb+16..kb+23}, kb=0 ; lanes 16-31: kb=8.
__device__ __forceinline__ bfvec16 load_b_frag(const unsigned short* buf, int n_off, int lane) {
    int n  = n_off + (lane & 15);
    int kb = (lane >> 4) * 8;
    bfvec8 lo = *(const bfvec8*)(buf + n * 32 + kb);
    bfvec8 hi = *(const bfvec8*)(buf + n * 32 + kb + 16);
    return __builtin_shufflevector(lo, hi, 0,1,2,3,4,5,6,7,8,9,10,11,12,13,14,15);
}

__device__ __forceinline__ v8f wmma_bf16(bfvec16 a, bfvec16 b, v8f c) {
    return __builtin_amdgcn_wmma_f32_16x16x32_bf16(false, a, false, b, (short)0, c, false, false);
}

// ---------- kernel 0: zero output + counters ----------
__global__ void zero_kernel(float* __restrict__ out, int n, int* __restrict__ counts) {
    for (int i = blockIdx.x * blockDim.x + threadIdx.x; i < n; i += gridDim.x * blockDim.x)
        out[i] = 0.0f;
    if (blockIdx.x == 0 && threadIdx.x < NE) counts[threadIdx.x] = 0;
}

// ---------- kernel 1: gate (one wave per token) ----------
__global__ __launch_bounds__(32)
void gate_kernel(const float* __restrict__ x, const float* __restrict__ gw,
                 const float* __restrict__ bias,
                 int* __restrict__ topk_idx, float* __restrict__ topk_w,
                 int* __restrict__ counts) {
    const int t = blockIdx.x, lane = threadIdx.x;
    float acc[NE];
    #pragma unroll
    for (int e = 0; e < NE; e++) acc[e] = 0.0f;
    const float* xr = x + (size_t)t * DIMD;
    for (int d = lane; d < DIMD; d += 32) {
        float xv = xr[d];
        #pragma unroll
        for (int e = 0; e < NE; e++) acc[e] += xv * gw[e * DIMD + d];
    }
    #pragma unroll
    for (int off = 16; off > 0; off >>= 1) {
        #pragma unroll
        for (int e = 0; e < NE; e++) acc[e] += __shfl_down(acc[e], off);
    }
    if (lane == 0) {
        float s[NE], sb[NE];
        #pragma unroll
        for (int e = 0; e < NE; e++) {
            s[e]  = fast_sigmoid(acc[e]);
            sb[e] = s[e] + bias[e];
        }
        // group scores (4 groups of 4), pick top-2 groups (lowest index on tie)
        float gs[4];
        #pragma unroll
        for (int g = 0; g < 4; g++) {
            float m = sb[g*4];
            #pragma unroll
            for (int j = 1; j < 4; j++) m = fmaxf(m, sb[g*4 + j]);
            gs[g] = m;
        }
        int g0 = 0;
        for (int g = 1; g < 4; g++) if (gs[g] > gs[g0]) g0 = g;
        int g1 = -1;
        for (int g = 0; g < 4; g++) if (g != g0 && (g1 < 0 || gs[g] > gs[g1])) g1 = g;
        // top-4 experts among allowed groups (lowest index on tie)
        bool used[NE];
        #pragma unroll
        for (int e = 0; e < NE; e++) used[e] = false;
        int   idx[TOPKD];
        float wv[TOPKD];
        float wsum = 0.0f;
        for (int j = 0; j < TOPKD; j++) {
            int best = -1;
            for (int e = 0; e < NE; e++) {
                int g = e >> 2;
                if (g != g0 && g != g1) continue;
                if (used[e]) continue;
                if (best < 0 || sb[e] > sb[best]) best = e;
            }
            idx[j] = best; used[best] = true;
            wv[j]  = s[best];
            wsum  += wv[j];
        }
        float inv = 1.0f / wsum;   // ROUTE_SCALE == 1
        #pragma unroll
        for (int j = 0; j < TOPKD; j++) {
            topk_idx[t*TOPKD + j] = idx[j];
            topk_w  [t*TOPKD + j] = wv[j] * inv;
            atomicAdd(&counts[idx[j]], 1);
        }
    }
}

// ---------- kernel 2: deterministic per-expert packed row lists ----------
__global__ __launch_bounds__(32)
void build_lists(const int* __restrict__ topk_idx, const float* __restrict__ topk_w,
                 const int* __restrict__ counts,
                 int* __restrict__ row_token, float* __restrict__ row_w) {
    const int e = blockIdx.x, lane = threadIdx.x;
    if (e >= NE) {   // shared pseudo-experts: identity lists, weight 1
        int base = ROWS_ROUTED + (e - NE) * TTOK;
        for (int t = lane; t < TTOK; t += 32) {
            row_token[base + t] = t;
            row_w[base + t] = 1.0f;
        }
        return;
    }
    int off = 0;
    for (int j = 0; j < e; j++) off += counts[j];
    for (int base = 0; base < TTOK; base += 32) {
        int t = base + lane;
        bool sel = false; float w = 0.0f;
        #pragma unroll
        for (int j = 0; j < TOPKD; j++) {
            if (topk_idx[t*TOPKD + j] == e) { sel = true; w = topk_w[t*TOPKD + j]; }
        }
        unsigned long long m = __ballot(sel);
        int rank = __popcll(m & ((1ull << lane) - 1ull));
        if (sel) { row_token[off + rank] = t; row_w[off + rank] = w; }
        off += __popcll(m);
    }
}

// ---------- kernel 3: H = silu(Xg*w1) * (Xg*w3), grouped GEMM ----------
__global__ __launch_bounds__(256)
void stage1_kernel(const float* __restrict__ x,
                   const float* __restrict__ w1, const float* __restrict__ w3,
                   const float* __restrict__ sw1, const float* __restrict__ sw3,
                   const int* __restrict__ counts, const int* __restrict__ row_token,
                   unsigned short* __restrict__ H) {
    __shared__ __align__(16) unsigned short Abuf[128 * 32];
    __shared__ __align__(16) unsigned short B1buf[128 * 32];
    __shared__ __align__(16) unsigned short B3buf[128 * 32];
    __shared__ int sm_tok[128];

    const int e = blockIdx.z;
    int cnt, rowbase; size_t ldb;
    const float *b1, *b3;
    if (e < NE) {
        cnt = counts[e];
        rowbase = 0;
        for (int j = 0; j < e; j++) rowbase += counts[j];
        b1 = w1 + (size_t)e * DIMD * INTERD;
        b3 = w3 + (size_t)e * DIMD * INTERD;
        ldb = INTERD;
    } else {
        cnt = TTOK;
        rowbase = ROWS_ROUTED + (e - NE) * TTOK;
        b1 = sw1 + (size_t)(e - NE) * INTERD;
        b3 = sw3 + (size_t)(e - NE) * INTERD;
        ldb = 2 * INTERD;
    }
    const int m0 = blockIdx.y * 128;
    if (m0 >= cnt) return;
    const int nb  = blockIdx.x * 128;      // column base in INTER
    const int tid = threadIdx.x, lane = tid & 31, wid = tid >> 5;
    const int wm = wid >> 1, wn = wid & 1; // 4x2 wave grid: 32 rows x 64 cols each

    if (tid < 128) {
        int i = m0 + tid;
        sm_tok[tid] = (i < cnt) ? row_token[rowbase + i] : -1;
    }
    __syncthreads();

    v8f c1[2][4], c3[2][4];
    #pragma unroll
    for (int a = 0; a < 2; a++)
        #pragma unroll
        for (int b = 0; b < 4; b++) { c1[a][b] = (v8f)0.0f; c3[a][b] = (v8f)0.0f; }

    for (int k0 = 0; k0 < DIMD; k0 += 32) {
        // A tile: gather x rows, fp32 -> bf16, contiguous 8B LDS stores
        for (int it = tid; it < 128 * 8; it += 256) {
            int r = it >> 3, c4 = (it & 7) * 4;
            float4 v = make_float4(0.f, 0.f, 0.f, 0.f);
            int t = sm_tok[r];
            if (t >= 0) v = *(const float4*)(x + (size_t)t * DIMD + k0 + c4);
            ushort4 o;
            o.x = f2bf(v.x); o.y = f2bf(v.y); o.z = f2bf(v.z); o.w = f2bf(v.w);
            *(ushort4*)&Abuf[r * 32 + c4] = o;
        }
        // B tiles: each thread reads a 4-deep column segment (coalesced across
        // lanes) and writes one contiguous 8B transposed LDS store.
        for (int it = tid; it < 128 * 8; it += 256) {
            int cc = it & 127, kq = (it >> 7) * 4;
            const float* p1 = b1 + (size_t)(k0 + kq) * ldb + nb + cc;
            const float* p3 = b3 + (size_t)(k0 + kq) * ldb + nb + cc;
            ushort4 o1, o3;
            o1.x = f2bf(p1[0]);       o1.y = f2bf(p1[ldb]);
            o1.z = f2bf(p1[2 * ldb]); o1.w = f2bf(p1[3 * ldb]);
            o3.x = f2bf(p3[0]);       o3.y = f2bf(p3[ldb]);
            o3.z = f2bf(p3[2 * ldb]); o3.w = f2bf(p3[3 * ldb]);
            *(ushort4*)&B1buf[cc * 32 + kq] = o1;
            *(ushort4*)&B3buf[cc * 32 + kq] = o3;
        }
        __syncthreads();

        bfvec16 a0 = load_a_frag(Abuf, wm * 32,      lane);
        bfvec16 a1 = load_a_frag(Abuf, wm * 32 + 16, lane);
        #pragma unroll
        for (int fn = 0; fn < 4; fn++) {
            bfvec16 bf1 = load_b_frag(B1buf, wn * 64 + fn * 16, lane);
            c1[0][fn] = wmma_bf16(a0, bf1, c1[0][fn]);
            c1[1][fn] = wmma_bf16(a1, bf1, c1[1][fn]);
            bfvec16 bf3 = load_b_frag(B3buf, wn * 64 + fn * 16, lane);
            c3[0][fn] = wmma_bf16(a0, bf3, c3[0][fn]);
            c3[1][fn] = wmma_bf16(a1, bf3, c3[1][fn]);
        }
        __syncthreads();
    }

    // epilogue: h = silu(c1)*c3 -> bf16 H (fast rcp instead of IEEE divide)
    const int lhalf = lane >> 4, lcol = lane & 15;
    #pragma unroll
    for (int fm = 0; fm < 2; fm++)
        #pragma unroll
        for (int fn = 0; fn < 4; fn++)
            #pragma unroll
            for (int v = 0; v < 8; v++) {
                int row = wm * 32 + fm * 16 + (lhalf ? v + 8 : v);
                if (m0 + row < cnt) {
                    int col = nb + wn * 64 + fn * 16 + lcol;
                    float h1 = c1[fm][fn][v], h3 = c3[fm][fn][v];
                    float h = h1 * fast_sigmoid(h1) * h3;
                    H[(size_t)(rowbase + m0 + row) * INTERD + col] = f2bf(h);
                }
            }
}

// ---------- kernel 4: Y += cw * (H * w2), scatter with atomics ----------
__global__ __launch_bounds__(256)
void stage2_kernel(const unsigned short* __restrict__ H,
                   const float* __restrict__ w2, const float* __restrict__ sw2,
                   const int* __restrict__ counts, const int* __restrict__ row_token,
                   const float* __restrict__ row_w, float* __restrict__ out) {
    __shared__ __align__(16) unsigned short Abuf[128 * 32];
    __shared__ __align__(16) unsigned short Bbuf[128 * 32];
    __shared__ int   sm_tok[128];
    __shared__ float sm_w[128];

    const int e = blockIdx.z;
    int cnt, rowbase;
    if (e < NE) {
        cnt = counts[e];
        rowbase = 0;
        for (int j = 0; j < e; j++) rowbase += counts[j];
    } else {
        cnt = TTOK;
        rowbase = ROWS_ROUTED + (e - NE) * TTOK;
    }
    const float* b2 = (e < NE) ? (w2  + (size_t)e        * INTERD * DIMD)
                               : (sw2 + (size_t)(e - NE) * INTERD * DIMD);
    const int m0 = blockIdx.y * 128;
    if (m0 >= cnt) return;
    const int nb  = blockIdx.x * 128;      // column base in DIM
    const int tid = threadIdx.x, lane = tid & 31, wid = tid >> 5;
    const int wm = wid >> 1, wn = wid & 1;

    if (tid < 128) {
        int i = m0 + tid;
        bool ok = (i < cnt);
        sm_tok[tid] = ok ? row_token[rowbase + i] : 0;
        sm_w[tid]   = ok ? row_w[rowbase + i] : 0.0f;
    }
    __syncthreads();

    v8f c[2][4];
    #pragma unroll
    for (int a = 0; a < 2; a++)
        #pragma unroll
        for (int b = 0; b < 4; b++) c[a][b] = (v8f)0.0f;

    for (int k0 = 0; k0 < INTERD; k0 += 32) {
        // A tile: H rows already bf16 -> straight copies to LDS.
        // Use the CDNA5 async global->LDS path when available (ASYNCcnt).
        for (int it = tid; it < 128 * 4; it += 256) {
            int r = it >> 2, c8 = (it & 3) * 8;
            const unsigned short* gsrc = H + (size_t)(rowbase + m0 + r) * INTERD + k0 + c8;
            unsigned short* ldst = &Abuf[r * 32 + c8];
#ifdef USE_ASYNC_LDS
            __builtin_amdgcn_global_load_async_to_lds_b128((v4i*)gsrc, (v4i*)ldst, 0, 0);
#else
            *(uint4*)ldst = *(const uint4*)gsrc;
#endif
        }
        // B tile: w2 fp32 -> bf16, transposed, contiguous 8B LDS stores
        for (int it = tid; it < 128 * 8; it += 256) {
            int cc = it & 127, kq = (it >> 7) * 4;
            const float* p = b2 + (size_t)(k0 + kq) * DIMD + nb + cc;
            ushort4 o;
            o.x = f2bf(p[0]);        o.y = f2bf(p[DIMD]);
            o.z = f2bf(p[2 * DIMD]); o.w = f2bf(p[3 * DIMD]);
            *(ushort4*)&Bbuf[cc * 32 + kq] = o;
        }
#ifdef USE_ASYNC_LDS
        __builtin_amdgcn_s_wait_asynccnt(0);
#endif
        __syncthreads();

        bfvec16 a0 = load_a_frag(Abuf, wm * 32,      lane);
        bfvec16 a1 = load_a_frag(Abuf, wm * 32 + 16, lane);
        #pragma unroll
        for (int fn = 0; fn < 4; fn++) {
            bfvec16 bf = load_b_frag(Bbuf, wn * 64 + fn * 16, lane);
            c[0][fn] = wmma_bf16(a0, bf, c[0][fn]);
            c[1][fn] = wmma_bf16(a1, bf, c[1][fn]);
        }
        __syncthreads();
    }

    const int lhalf = lane >> 4, lcol = lane & 15;
    #pragma unroll
    for (int fm = 0; fm < 2; fm++)
        #pragma unroll
        for (int fn = 0; fn < 4; fn++)
            #pragma unroll
            for (int v = 0; v < 8; v++) {
                int row = wm * 32 + fm * 16 + (lhalf ? v + 8 : v);
                if (m0 + row < cnt) {
                    int col = nb + wn * 64 + fn * 16 + lcol;
                    float val = c[fm][fn][v] * sm_w[row];
                    atomicAdd(&out[(size_t)sm_tok[row] * DIMD + col], val);
                }
            }
}

// ---------- host launcher ----------
extern "C" void kernel_launch(void* const* d_in, const int* in_sizes, int n_in,
                              void* d_out, int out_size, void* d_ws, size_t ws_size,
                              hipStream_t stream) {
    const float* x      = (const float*)d_in[0];
    const float* gate_w = (const float*)d_in[1];
    const float* bias   = (const float*)d_in[2];
    const float* w1     = (const float*)d_in[3];
    const float* w2     = (const float*)d_in[4];
    const float* w3     = (const float*)d_in[5];
    const float* sw1    = (const float*)d_in[6];
    const float* sw2    = (const float*)d_in[7];
    const float* sw3    = (const float*)d_in[8];
    float* out = (float*)d_out;

    // workspace layout
    char* ws = (char*)d_ws;
    int*   counts    = (int*)ws;                               ws += 256;
    int*   topk_idx  = (int*)ws;                               ws += TTOK * TOPKD * sizeof(int);
    float* topk_w    = (float*)ws;                             ws += TTOK * TOPKD * sizeof(float);
    int*   row_token = (int*)ws;                               ws += ROWS_TOTAL * sizeof(int);
    float* row_w     = (float*)ws;                             ws += ROWS_TOTAL * sizeof(float);
    ws = (char*)(((size_t)ws + 255) & ~(size_t)255);
    unsigned short* H = (unsigned short*)ws;                   // ROWS_TOTAL * INTERD bf16 (~33 MB)

    zero_kernel<<<4096, 256, 0, stream>>>(out, TTOK * DIMD, counts);
    gate_kernel<<<TTOK, 32, 0, stream>>>(x, gate_w, bias, topk_idx, topk_w, counts);
    build_lists<<<NPSE, 32, 0, stream>>>(topk_idx, topk_w, counts, row_token, row_w);
    stage1_kernel<<<dim3(INTERD / 128, TTOK / 128, NPSE), 256, 0, stream>>>(
        x, w1, w3, sw1, sw3, counts, row_token, H);
    stage2_kernel<<<dim3(DIMD / 128, TTOK / 128, NPSE), 256, 0, stream>>>(
        H, w2, sw2, counts, row_token, row_w, out);
}
